// PointNetSetAbstraction_9354438770946
// MI455X (gfx1250) — matrix operand order repaired and checked by reading, not compiled, not run on hardware
//
#include <hip/hip_runtime.h>
#include <hip/hip_bf16.h>

// ---- problem constants --------------------------------------------------
#define BB   2
#define CC   29
#define NN   8192
#define SS   1024
#define KK   16
#define DD   32           // C + 3

typedef __attribute__((ext_vector_type(16))) _Float16 v16h;
typedef __attribute__((ext_vector_type(8)))  _Float16 v8h;
typedef __attribute__((ext_vector_type(8)))  float    v8f;

// ---- wave32 butterfly reductions ---------------------------------------
__device__ __forceinline__ void waveArgMax(float& v, int& i) {
#pragma unroll
  for (int m = 16; m > 0; m >>= 1) {
    float ov = __shfl_xor(v, m, 32);
    int   oi = __shfl_xor(i, m, 32);
    if (ov > v || (ov == v && oi < i)) { v = ov; i = oi; }
  }
}
__device__ __forceinline__ void waveArgMin(float& v, int& i) {
#pragma unroll
  for (int m = 16; m > 0; m >>= 1) {
    float ov = __shfl_xor(v, m, 32);
    int   oi = __shfl_xor(i, m, 32);
    if (ov < v || (ov == v && oi < i)) { v = ov; i = oi; }
  }
}

// =========================================================================
// Kernel A: farthest point sampling. One workgroup per batch; the whole
// 8192-point cloud lives in LDS (96KB of the WGP's 320KB).
// =========================================================================
__global__ void __launch_bounds__(1024)
fps_kernel(const float* __restrict__ xyz,     // (B,3,N)
           float* __restrict__ nxz_ws,        // (B,3,S) workspace copy
           int*   __restrict__ fps_ws,        // (B,S)
           float* __restrict__ out_newxyz,    // (B,3,S) -> d_out[0]
           int*   __restrict__ out_fpsidx)    // (B,S) int bits -> d_out tail
{
  __shared__ float sx[NN], sy[NN], sz[NN];
  __shared__ float rv[32];
  __shared__ int   ri[32];
  __shared__ int   sidx[SS];
  __shared__ int   s_far;

  const int b   = blockIdx.x;
  const int tid = threadIdx.x;
  const int lane = tid & 31;
  const int wid  = tid >> 5;

  for (int p = tid; p < NN; p += 1024) {
    sx[p] = xyz[b * 3 * NN + 0 * NN + p];
    sy[p] = xyz[b * 3 * NN + 1 * NN + p];
    sz[p] = xyz[b * 3 * NN + 2 * NN + p];
  }
  if (tid == 0) s_far = 0;

  float dist[8];
#pragma unroll
  for (int j = 0; j < 8; ++j) dist[j] = 1.0e10f;
  __syncthreads();

  for (int s = 0; s < SS; ++s) {
    const int far = s_far;
    if (tid == 0) sidx[s] = far;
    const float cx = sx[far], cy = sy[far], cz = sz[far];

    float lv = -1.0f; int li = 0;
#pragma unroll
    for (int j = 0; j < 8; ++j) {
      const int p = j * 1024 + tid;
      const float dx = sx[p] - cx, dy = sy[p] - cy, dz = sz[p] - cz;
      float d = dx * dx + dy * dy + dz * dz;
      d = fminf(dist[j], d);
      dist[j] = d;
      if (d > lv || (d == lv && p < li)) { lv = d; li = p; }
    }
    waveArgMax(lv, li);
    if (lane == 0) { rv[wid] = lv; ri[wid] = li; }
    __syncthreads();
    if (wid == 0) {
      float v = rv[lane]; int i = ri[lane];
      waveArgMax(v, i);
      if (lane == 0) s_far = i;
    }
    __syncthreads();
  }

  // emit indices + gathered coordinates
  {
    const int idx = sidx[tid];
    fps_ws[b * SS + tid]     = idx;
    out_fpsidx[b * SS + tid] = idx;
  }
  for (int t = tid; t < 3 * SS; t += 1024) {
    const int d = t >> 10, s2 = t & (SS - 1);
    const float val = xyz[b * 3 * NN + d * NN + sidx[s2]];
    nxz_ws[b * 3 * SS + t]     = val;
    out_newxyz[b * 3 * SS + t] = val;
  }
}

// =========================================================================
// Kernel B: KNN among the S sampled points. One wave per query; 32
// distances per lane staged in LDS; 16 rounds of wave argmin.
// =========================================================================
__global__ void __launch_bounds__(128)
knn_kernel(const float* __restrict__ nxz,     // (B,3,S)
           int* __restrict__ knn_ws)          // (B*S,K)
{
  __shared__ float dmat[4][32][32];
  const int lane = threadIdx.x & 31;
  const int w    = threadIdx.x >> 5;
  const int q    = blockIdx.x * 4 + w;
  const int b    = q >> 10;
  const int s    = q & (SS - 1);

  const float qx = nxz[b * 3 * SS + 0 * SS + s];
  const float qy = nxz[b * 3 * SS + 1 * SS + s];
  const float qz = nxz[b * 3 * SS + 2 * SS + s];

#pragma unroll 4
  for (int t = 0; t < 32; ++t) {
    const int j = t * 32 + lane;
    const float dx = nxz[b * 3 * SS + 0 * SS + j] - qx;
    const float dy = nxz[b * 3 * SS + 1 * SS + j] - qy;
    const float dz = nxz[b * 3 * SS + 2 * SS + j] - qz;
    dmat[w][t][lane] = dx * dx + dy * dy + dz * dz;
  }

  for (int kk = 0; kk < KK; ++kk) {
    float bv = 3.0e38f; int bj = 0x7fffffff;
#pragma unroll 4
    for (int t = 0; t < 32; ++t) {
      const float d = dmat[w][t][lane];
      const int j = t * 32 + lane;
      if (d < bv || (d == bv && j < bj)) { bv = d; bj = j; }
    }
    waveArgMin(bv, bj);                // all lanes hold the winner
    if (lane == (bj & 31)) dmat[w][bj >> 5][lane] = 3.0e38f;
    if (lane == 0) knn_ws[q * KK + kk] = bj;
  }
}

// =========================================================================
// Kernel C: repack conv weights to f16 in WMMA-fragment order.
//
// Target layout per layer: tiles of (M-tile mt, K-block kb); each tile is
// 512 halfs = 32 lanes x 16 halfs, lane-contiguous. GEMM K dimension is
// reordered to (kh,kw,ci) so B-fragments are channel-contiguous in LDS.
//
// Fragment element (lane,e) holds A[co][kidx]:
//   co   = mt*16 + (lane&15)
//   koff = (lane<16) ? 0 : 16
//   kidx = kb*32 + koff + e                              (A 16x32 layout)
//   kidx = spk*Cin + ci ;  source = co*Ktot + ci*ksz2 + spk   (OIHW)
// =========================================================================
#define W0N 147456   // 64 x2304 : nkb=72, nmt=4
#define W1N 36864    // 64 x 576 : nkb=18, nmt=4
#define W2N 73728    // 128x 576 : nkb=18, nmt=8
#define W3N 65536    // 128x 512 : nkb=16, nmt=8
#define WTN (W0N + W1N + W2N + W3N)

__device__ __forceinline__ int pack_src(int t, int Ktot, int nkb, int Cin, int ksz2) {
  const int tile = t >> 9;           // /512
  const int u    = t & 511;
  const int lane = u >> 4;
  const int e    = u & 15;
  const int mt   = tile / nkb;
  const int kb   = tile - mt * nkb;
  const int co   = mt * 16 + (lane & 15);
  const int koff = (lane < 16) ? 0 : 16;
  const int kidx = kb * 32 + koff + e;
  const int spk  = kidx / Cin;
  const int ci   = kidx - spk * Cin;
  return co * Ktot + ci * ksz2 + spk;
}

__global__ void __launch_bounds__(256)
w2h_kernel(const float* __restrict__ w0, const float* __restrict__ w1,
           const float* __restrict__ w2, const float* __restrict__ w3,
           _Float16* __restrict__ out)
{
  const int t = blockIdx.x * 256 + threadIdx.x;
  if (t >= WTN) return;
  float v;
  if (t < W0N) {
    v = w0[pack_src(t, 2304, 72, 256, 9)];
  } else if (t < W0N + W1N) {
    v = w1[pack_src(t - W0N, 576, 18, 64, 9)];
  } else if (t < W0N + W1N + W2N) {
    v = w2[pack_src(t - W0N - W1N, 576, 18, 64, 9)];
  } else {
    v = w3[pack_src(t - W0N - W1N - W2N, 512, 16, 128, 4)];
  }
  out[t] = (_Float16)v;
}

// =========================================================================
// WMMA GEMM: A pre-swizzled in global (one v16h load / lane / K-block),
// B gathered as 16 channel-contiguous halfs from spatial-major LDS
// activations (one v16h load = two ds_load_b128). Invalid N columns read a
// zeroed LDS row via a single pointer select.
// Cin / ksz / W are literals at every call site (forceinline -> folded).
// =========================================================================
__device__ __forceinline__ void
gemm_tile(const _Float16* __restrict__ Apack,  // layer base + mt*nkb*512
          int nkb, int Cin, int ksz, int W,
          const _Float16* Xt,                  // LDS [sp][Cin]
          const _Float16* zrow,                // LDS zero row (>=32 halfs)
          int n_sp, bool nvalid, int lane, v8f& acc)
{
  const int kb0 = (lane < 16) ? 0 : 16;
  for (int kb = 0; kb < nkb; ++kb) {
    const v16h a = *reinterpret_cast<const v16h*>(Apack + kb * 512 + lane * 16);
    const int kk  = kb * 32;
    const int spk = kk / Cin;                          // uniform, folded
    const int ksp = (spk / ksz) * W + (spk % ksz);     // input-row offset
    const int cia = (kk % Cin) + kb0;
    const _Float16* p = nvalid ? (Xt + (n_sp + ksp) * Cin + cia) : zrow;
    const v16h bv = *reinterpret_cast<const v16h*>(p);
    acc = __builtin_amdgcn_wmma_f32_16x16x32_f16(
        false, a, false, bv, (short)0, acc, false, false);
  }
}

// =========================================================================
// Kernel D: fused group -> outer -> squeeze_2x2 -> conv0..conv3 (+ReLU).
// One 128-thread workgroup (4 waves) per sample; activations live in LDS
// in spatial-major [sp][Cin] order; outputs written back as v8h fragments.
// =========================================================================
__global__ void __launch_bounds__(128)
fused_kernel(const float* __restrict__ nxz,      // (B,3,S)
             const float* __restrict__ feats,    // (B,C,N)
             const int*   __restrict__ knn_ws,   // (B*S,K)
             const _Float16* __restrict__ wh,    // packed f16 weights
             float* __restrict__ out_np)         // (B,128,S)
{
  __shared__ __align__(32) _Float16 X0t[64 * 256];  // conv0 in : sp 8x8, 256ch
  __shared__ __align__(32) _Float16 X1t[36 * 64];   // conv0 out: sp 6x6, 64ch
  __shared__ __align__(32) _Float16 X2t[16 * 64];   // conv1 out: sp 4x4, 64ch
  __shared__ __align__(32) _Float16 X3t[4 * 128];   // conv2 out: sp 2x2, 128ch
  __shared__ __align__(32) _Float16 zrow[32];       // zero row for invalid N
  __shared__ float    fgL[DD * KK];                 // grouped feats (32 x 16)
  __shared__ int      nidx[KK];

  const int tid  = threadIdx.x;
  const int lane = tid & 31;
  const int w    = tid >> 5;
  const int hi8  = (lane < 16) ? 0 : 8;
  const int sample = blockIdx.x;
  const int b = sample >> 10;
  const int s = sample & (SS - 1);

  const _Float16* A0 = wh;
  const _Float16* A1 = wh + W0N;
  const _Float16* A2 = wh + W0N + W1N;
  const _Float16* A3 = wh + W0N + W1N + W2N;

  if (tid < KK) nidx[tid] = knn_ws[sample * KK + tid];
  if (tid < 32) zrow[tid] = (_Float16)0;
  __syncthreads();

  // fg = concat(pos_diff, grouped feats): (32, 16)
  for (int t = tid; t < DD * KK; t += 128) {
    const int d = t >> 4, i = t & 15;
    const int j = nidx[i];
    float val;
    if (d < 3)
      val = nxz[b * 3 * SS + d * SS + j] - nxz[b * 3 * SS + d * SS + s];
    else
      val = feats[(size_t)b * CC * NN + (size_t)(d - 3) * NN + j];
    fgL[d * KK + i] = val;
  }
  __syncthreads();

  // X0t: outer product + ones, squeeze_2x2, stored spatial-major [sp][256]
  for (int t = tid; t < 64 * 256; t += 128) {
    const int sp = t >> 8, ch = t & 255;
    const int oh = sp >> 3, ow = sp & 7;
    const int cq = ch >> 6, c = ch & 63;
    const int rh = (cq == 1 || cq == 3);
    const int rw = (cq == 1 || cq == 2);
    float v = 1.0f;
    if (c < 32) v = fgL[c * KK + (2 * oh + rh)] * fgL[c * KK + (2 * ow + rw)];
    X0t[sp * 256 + ch] = (_Float16)v;
  }
  __syncthreads();

  // ---- conv0: M=64 (1 tile/wave), N=36 (3 tiles), K=2304 ----
  {
    const int mbase = w * 16;
    const _Float16* Ap = A0 + (size_t)w * 72 * 512;
#pragma unroll
    for (int nb = 0; nb < 48; nb += 16) {
      const int n = nb + (lane & 15);
      const bool nvalid = n < 36;
      const int n_sp = nvalid ? ((n / 6) * 8 + (n % 6)) : 0;
      v8f acc = {};
      gemm_tile(Ap, 72, 256, 3, 8, X0t, zrow, n_sp, nvalid, lane, acc);
      v8h st;
#pragma unroll
      for (int r = 0; r < 8; ++r) st[r] = (_Float16)fmaxf(acc[r], 0.0f);
      if (nvalid) *reinterpret_cast<v8h*>(X1t + n * 64 + mbase + hi8) = st;
    }
  }
  __syncthreads();

  // ---- conv1: M=64, N=16 (4x4), K=576 ----
  {
    const int mbase = w * 16;
    const _Float16* Ap = A1 + (size_t)w * 18 * 512;
    const int n = lane & 15;
    const int n_sp = (n >> 2) * 6 + (n & 3);
    v8f acc = {};
    gemm_tile(Ap, 18, 64, 3, 6, X1t, zrow, n_sp, true, lane, acc);
    v8h st;
#pragma unroll
    for (int r = 0; r < 8; ++r) st[r] = (_Float16)fmaxf(acc[r], 0.0f);
    *reinterpret_cast<v8h*>(X2t + n * 64 + mbase + hi8) = st;
  }
  __syncthreads();

  // ---- conv2: M=128 (2 tiles/wave), N=4 valid of 16, K=576 ----
#pragma unroll
  for (int half = 0; half < 2; ++half) {
    const int mt = w + half * 4;
    const int mbase = mt * 16;
    const _Float16* Ap = A2 + (size_t)mt * 18 * 512;
    const int n = lane & 15;
    const bool nvalid = n < 4;
    const int n_sp = nvalid ? ((n >> 1) * 4 + (n & 1)) : 0;
    v8f acc = {};
    gemm_tile(Ap, 18, 64, 3, 4, X2t, zrow, n_sp, nvalid, lane, acc);
    v8h st;
#pragma unroll
    for (int r = 0; r < 8; ++r) st[r] = (_Float16)fmaxf(acc[r], 0.0f);
    if (nvalid) *reinterpret_cast<v8h*>(X3t + n * 128 + mbase + hi8) = st;
  }
  __syncthreads();

  // ---- conv3: M=128 (2 tiles/wave), N=1 valid, K=512 ----
#pragma unroll
  for (int half = 0; half < 2; ++half) {
    const int mt = w + half * 4;
    const int mbase = mt * 16;
    const _Float16* Ap = A3 + (size_t)mt * 16 * 512;
    const bool nvalid = (lane & 15) == 0;
    v8f acc = {};
    gemm_tile(Ap, 16, 128, 2, 2, X3t, zrow, 0, nvalid, lane, acc);
#pragma unroll
    for (int r = 0; r < 8; ++r) {
      const int co = mbase + r + hi8;
      if (nvalid) out_np[(size_t)b * 128 * SS + (size_t)co * SS + s] =
          fmaxf(acc[r], 0.0f);
    }
  }
}

// =========================================================================
// Launch
// =========================================================================
extern "C" void kernel_launch(void* const* d_in, const int* in_sizes, int n_in,
                              void* d_out, int out_size, void* d_ws, size_t ws_size,
                              hipStream_t stream) {
  const float* xyz   = (const float*)d_in[0];
  const float* feats = (const float*)d_in[1];
  const float* w0    = (const float*)d_in[2];
  const float* w1    = (const float*)d_in[3];
  const float* w2    = (const float*)d_in[4];
  const float* w3    = (const float*)d_in[5];

  // workspace layout
  char* ws = (char*)d_ws;
  int*      fps_ws = (int*)ws;                     //  8 KB
  float*    nxz_ws = (float*)(ws + 8192);          // 24 KB
  int*      knn_ws = (int*)(ws + 32768);           // 128 KB
  _Float16* wh     = (_Float16*)(ws + 163840);     // 632 KB

  // output layout: new_xyz (B*3*S) | new_points (B*128*S) | fps_idx (B*S, i32 bits)
  float* out_newxyz = (float*)d_out;
  float* out_np     = (float*)d_out + BB * 3 * SS;
  int*   out_idx    = (int*)((float*)d_out + BB * 3 * SS + BB * 128 * SS);

  fps_kernel<<<BB, 1024, 0, stream>>>(xyz, nxz_ws, fps_ws, out_newxyz, out_idx);
  knn_kernel<<<(BB * SS) / 4, 128, 0, stream>>>(nxz_ws, knn_ws);
  w2h_kernel<<<(WTN + 255) / 256, 256, 0, stream>>>(w0, w1, w2, w3, wh);
  fused_kernel<<<BB * SS, 128, 0, stream>>>(nxz_ws, feats, knn_ws, wh, out_np);
}
